// BinaryLinear_79714593014187
// MI455X (gfx1250) — 8-bit matrix operand order repaired and checked by reading, compiled, not run on hardware
//
#include <hip/hip_runtime.h>

typedef int v8i __attribute__((ext_vector_type(8)));

#define TOKENS 32768
#define IN_F   512
#define OUT_F  512
#define MBLK   32
#define KSTEP  64

__global__ __launch_bounds__(256) void quantize_weight_kernel(
    const float* __restrict__ w, const float* __restrict__ thr,
    char* __restrict__ wq) {
  int t = blockIdx.x * 256 + threadIdx.x;
  int o = t >> 7;
  int k = (t & 127) << 2;
  const float4 v = *(const float4*)(w + (size_t)o * IN_F + k);
  const float  th = thr[o];
  char4 q;
  q.x = ((v.x - th) >= 0.0f) ? (char)1 : (char)-1;
  q.y = ((v.y - th) >= 0.0f) ? (char)1 : (char)-1;
  q.z = ((v.z - th) >= 0.0f) ? (char)1 : (char)-1;
  q.w = ((v.w - th) >= 0.0f) ? (char)1 : (char)-1;
  *(char4*)(wq + (size_t)o * IN_F + k) = q;
}

__global__ __launch_bounds__(256) void bin_gemm_iu8_kernel(
    const float* __restrict__ x, const char* __restrict__ wq,
    const float* __restrict__ shiftp, float* __restrict__ out) {
  __shared__ uint2 ldsA[(MBLK * KSTEP) / 8];

  const int tid  = threadIdx.x;
  const int wave = tid >> 5;
  const int lane = tid & 31;
  const int half = lane >> 4;
  const int l16  = lane & 15;
  const int m0   = blockIdx.x * MBLK;

  float sh = fminf(fmaxf(shiftp[0], -8.0f), 0.0f);
  const float scale = exp2f(rintf(sh));

  const v8i vzero = {};
  v8i acc[2][4];
#pragma unroll
  for (int mt = 0; mt < 2; ++mt)
#pragma unroll
    for (int nt = 0; nt < 4; ++nt) acc[mt][nt] = vzero;

  const int srow = tid >> 3;
  const int scol = (tid & 7) << 3;
  const float* xrow = x + (size_t)(m0 + srow) * IN_F + scol;

  for (int kk = 0; kk < IN_F; kk += KSTEP) {
    __syncthreads();

    float4 f0 = *(const float4*)(xrow + kk);
    float4 f1 = *(const float4*)(xrow + kk + 4);
    if (kk + KSTEP < IN_F)
      __builtin_prefetch(xrow + kk + KSTEP, 0, 0);
    unsigned int p0 = (f0.x >= 0.0f ? 0x01u : 0xFFu)
                    | (f0.y >= 0.0f ? 0x01u : 0xFFu) << 8
                    | (f0.z >= 0.0f ? 0x01u : 0xFFu) << 16
                    | (f0.w >= 0.0f ? 0x01u : 0xFFu) << 24;
    unsigned int p1 = (f1.x >= 0.0f ? 0x01u : 0xFFu)
                    | (f1.y >= 0.0f ? 0x01u : 0xFFu) << 8
                    | (f1.z >= 0.0f ? 0x01u : 0xFFu) << 16
                    | (f1.w >= 0.0f ? 0x01u : 0xFFu) << 24;
    ldsA[tid] = make_uint2(p0, p1);
    __syncthreads();

    v8i afrag[2];
#pragma unroll
    for (int mt = 0; mt < 2; ++mt) {
      const int base = (mt * 16 + l16) * 8 + half * 2;
#pragma unroll
      for (int p = 0; p < 4; ++p) {
        uint2 q = ldsA[base + (p & 1) + (p >> 1) * 4];
        afrag[mt][2 * p]     = (int)q.x;
        afrag[mt][2 * p + 1] = (int)q.y;
      }
    }

#pragma unroll
    for (int nt = 0; nt < 4; ++nt) {
      const int o = wave * 64 + nt * 16 + l16;
      const char* bp = wq + (size_t)o * IN_F + kk + half * 16;
      int4 b0 = *(const int4*)(bp);
      int4 b1 = *(const int4*)(bp + 32);
      v8i bfrag;
      bfrag[0] = b0.x; bfrag[1] = b0.y; bfrag[2] = b0.z; bfrag[3] = b0.w;
      bfrag[4] = b1.x; bfrag[5] = b1.y; bfrag[6] = b1.z; bfrag[7] = b1.w;
#pragma unroll
      for (int mt = 0; mt < 2; ++mt) {
        acc[mt][nt] = __builtin_amdgcn_wmma_i32_16x16x64_iu8(
             true, afrag[mt],  true, bfrag,
            acc[mt][nt],  false,  false);
      }
    }
  }

#pragma unroll
  for (int mt = 0; mt < 2; ++mt) {
#pragma unroll
    for (int nt = 0; nt < 4; ++nt) {
      const int n = wave * 64 + nt * 16 + l16;
#pragma unroll
      for (int r = 0; r < 8; ++r) {
        const int m = m0 + mt * 16 + half * 8 + r;
        out[(size_t)m * OUT_F + n] = scale * (float)acc[mt][nt][r];
      }
    }
  }
}

extern "C" void kernel_launch(void* const* d_in, const int* in_sizes, int n_in,
                              void* d_out, int out_size, void* d_ws, size_t ws_size,
                              hipStream_t stream) {
  const float* x     = (const float*)d_in[0];
  const float* w     = (const float*)d_in[1];
  const float* thr   = (const float*)d_in[2];
  const float* shift = (const float*)d_in[3];
  float* out = (float*)d_out;
  char*  wq  = (char*)d_ws;

  quantize_weight_kernel<<<(OUT_F * IN_F) / (256 * 4), 256, 0, stream>>>(w, thr, wq);
  bin_gemm_iu8_kernel<<<TOKENS / MBLK, 256, 0, stream>>>(x, wq, shift, out);
}
